// SimpleRNNClassifier_61246233640989
// MI455X (gfx1250) — matrix operand order repaired and checked by reading, not compiled, not run on hardware
//
#include <hip/hip_runtime.h>
#include <hip/hip_bf16.h>
#include <math.h>

typedef __attribute__((ext_vector_type(16))) _Float16 v16h;
typedef __attribute__((ext_vector_type(8)))  _Float16 v8h;
typedef __attribute__((ext_vector_type(8)))  float    v8f;

#define S_LEN 512
#define B_SZ  1024
#define E_DIM 64
#define H_DIM 64
#define NTILE 4                 // 64 output cols / 16
#define TILE_HALVES 1024        // 4 n-tiles * 32 lanes * 8 halves per (row-tile, s)

static __device__ __forceinline__ v8f wmma_f16(v16h a, v16h b, v8f c) {
    return __builtin_amdgcn_wmma_f32_16x16x32_f16(
        /*neg_a=*/false, a, /*neg_b=*/false, b,
        /*c_mod=*/(short)0, c, /*reuse_a=*/false, /*reuse_b=*/false);
}

static __device__ __forceinline__ float fast_exp2(float x) {
#if __has_builtin(__builtin_amdgcn_exp2f)
    return __builtin_amdgcn_exp2f(x);
#else
    return exp2f(x);
#endif
}

// gfx1250 has a hardware v_tanh_f32 (TRANS op).  Fall back to a saturating
// exp2/rcp form (4 VALU ops) if the builtin is absent in this toolchain.
static __device__ __forceinline__ float fast_tanh(float x) {
#if __has_builtin(__builtin_amdgcn_tanhf)
    return __builtin_amdgcn_tanhf(x);
#elif __has_builtin(__builtin_amdgcn_tanh_f32)
    return __builtin_amdgcn_tanh_f32(x);
#else
    const float e = fast_exp2(x * 2.885390081777927f);   // e^(2x)
    return 1.0f - 2.0f * __builtin_amdgcn_rcpf(e + 1.0f);
#endif
}

static __device__ __forceinline__ float fast_sigmoid(float x) {
    return __builtin_amdgcn_rcpf(1.0f + fast_exp2(-x * 1.4426950408889634f));
}

// Build a B-fragment (32x16, f16) for D = A x B with B[k][n] = W[n][k]
// (i.e. multiply by W^T).  Lane holds column N = lane%16; half-index i maps
// to K = (i&7) + (lane<16 ? 0 : 8) + (i>=8 ? 16 : 0) within the 32-wide K slice starting at k0.
static __device__ __forceinline__ v16h load_b_wT(const float* __restrict__ W,
                                                 int n0, int k0, int r, int hi) {
    const float* wrow = W + (n0 + r) * H_DIM + k0;
    v16h b;
#pragma unroll
    for (int i = 0; i < 16; ++i) b[i] = (_Float16)wrow[(i & 7) + hi * 8 + ((i >= 8) ? 16 : 0)];
    return b;
}

// ---------------------------------------------------------------------------
// Phase 1: embedding gather + x_proj = emb[x] @ W_ih^T + (b_ih + b_hh),
// written as f16 C-fragment tiles: xws[(b_tile*512+s)*4 + n][lane][8].
// Embedding rows are gathered into LDS with GLOBAL_LOAD_ASYNC_TO_LDS_B128
// (no VGPR staging), completion via s_wait_asynccnt.
// ---------------------------------------------------------------------------
__global__ __launch_bounds__(32)
void rnn_xproj_kernel(const int*   __restrict__ x,      // [B,S] token ids
                      const float* __restrict__ emb,    // [VOCAB,64]
                      const float* __restrict__ W_ih,   // [64,64]
                      const float* __restrict__ b_ih,   // [64]
                      const float* __restrict__ b_hh,   // [64]
                      _Float16*    __restrict__ xws)    // [64*512*4*32*8]
{
    __shared__ float lds_emb[16][E_DIM];   // only LDS object -> offset 0

    const int t      = blockIdx.x;          // = b_tile*512 + s
    const int s      = t & (S_LEN - 1);
    const int b_tile = t >> 9;
    const int lane   = threadIdx.x;
    const int r      = lane & 15;
    const int hi     = lane >> 4;

    // Async-gather 16 embedding rows (16 x 256B) into LDS.
    // Iteration `it`: lanes 0-15 copy 16B chunks of row 2*it, lanes 16-31
    // of row 2*it+1.  LDS byte offset = row*256 + r*16 (array base offset 0).
#pragma unroll
    for (int it = 0; it < 8; ++it) {
        const int row  = it * 2 + hi;
        const int tok  = x[(b_tile * 16 + row) * S_LEN + s];
        const float* gp = emb + (size_t)tok * E_DIM + r * 4;  // 16B chunk
        const unsigned lds_off = (unsigned)(row * 256 + r * 16);
        asm volatile("global_load_async_to_lds_b128 %0, %1, off"
                     :: "v"(lds_off), "v"(gp) : "memory");
    }
#if __has_builtin(__builtin_amdgcn_s_wait_asynccnt)
    __builtin_amdgcn_s_wait_asynccnt(0);
#else
    asm volatile("s_wait_asynccnt 0x0" ::: "memory");
#endif
    __syncthreads();

    // A-fragments (16x32 f16) for K-slices e=0..31 and e=32..63.
    v16h a0, a1;
#pragma unroll
    for (int i = 0; i < 16; ++i) {
        const int kl = (i & 7) + hi * 8 + ((i >= 8) ? 16 : 0);
        a0[i] = (_Float16)lds_emb[r][kl];
        a1[i] = (_Float16)lds_emb[r][32 + kl];
    }

    // C init = bias broadcast down each column N = n*16 + r.
    v8f c[NTILE];
#pragma unroll
    for (int n = 0; n < NTILE; ++n) {
        const float bv = b_ih[n * 16 + r] + b_hh[n * 16 + r];
#pragma unroll
        for (int j = 0; j < 8; ++j) c[n][j] = bv;
    }

#pragma unroll
    for (int n = 0; n < NTILE; ++n) {
        const v16h b0 = load_b_wT(W_ih, n * 16, 0,  r, hi);
        const v16h b1 = load_b_wT(W_ih, n * 16, 32, r, hi);
        c[n] = wmma_f16(a0, b0, c[n]);
        c[n] = wmma_f16(a1, b1, c[n]);
    }

    // Store C-fragments as f16, one 128-bit store per lane per n-tile.
#pragma unroll
    for (int n = 0; n < NTILE; ++n) {
        v8h o;
#pragma unroll
        for (int j = 0; j < 8; ++j) o[j] = (_Float16)c[n][j];
        *(v8h*)(xws + (((size_t)t * NTILE + n) * 32 + lane) * 8) = o;
    }
}

// ---------------------------------------------------------------------------
// Phase 2: recurrent scan.  64 blocks x 1 wave; block b_tile owns 16 batch
// rows.  h lives in C-fragment registers; W_hh lives in B-fragment registers
// for all 512 steps; per step: h -> LDS -> A-fragment, c = x_proj (streamed
// f16 C-fragments), 8 WMMAs, hardware tanh.  Then FC + sigmoid.
// ---------------------------------------------------------------------------
__global__ __launch_bounds__(32)
void rnn_scan_kernel(const _Float16* __restrict__ xws,  // phase-1 output
                     const float*    __restrict__ W_hh, // [64,64]
                     const float*    __restrict__ fc_w, // [64]
                     const float*    __restrict__ fc_b, // [1]
                     float*          __restrict__ out)  // [1024]
{
    __shared__ _Float16 lds_h[16][H_DIM];
    __shared__ float    lds_hf[16][H_DIM + 1];

    const int b_tile = blockIdx.x;
    const int lane   = threadIdx.x;
    const int r      = lane & 15;
    const int hi     = lane >> 4;

    // W_hh^T as resident B-fragments (4 n-tiles x 2 K-slices).
    v16h bw[NTILE][2];
#pragma unroll
    for (int n = 0; n < NTILE; ++n) {
        bw[n][0] = load_b_wT(W_hh, n * 16, 0,  r, hi);
        bw[n][1] = load_b_wT(W_hh, n * 16, 32, r, hi);
    }

    v8f h[NTILE];
#pragma unroll
    for (int n = 0; n < NTILE; ++n)
#pragma unroll
        for (int j = 0; j < 8; ++j) h[n][j] = 0.0f;

    const _Float16* xp_base = xws + (size_t)b_tile * S_LEN * TILE_HALVES;

    for (int s = 0; s < S_LEN; ++s) {
        // Stream-ahead hint for the x_proj tiles (global_prefetch_b8).
        {
            const int sp = (s + 8 < S_LEN) ? (s + 8) : s;
            __builtin_prefetch(xp_base + (size_t)sp * TILE_HALVES + lane * 8, 0, 1);
        }

        // h (C layout: col = n*16+r, row = j + hi*8) -> LDS row-major f16.
#pragma unroll
        for (int n = 0; n < NTILE; ++n) {
            const int col = n * 16 + r;
#pragma unroll
            for (int j = 0; j < 8; ++j)
                lds_h[j + hi * 8][col] = (_Float16)h[n][j];
        }
        __syncthreads();

        // Rebuild A-fragments of h from LDS (contiguous b128 reads).
        v16h a0, a1;
#pragma unroll
        for (int i = 0; i < 16; ++i) {
            const int kl = (i & 7) + hi * 8 + ((i >= 8) ? 16 : 0);
            a0[i] = lds_h[r][kl];
            a1[i] = lds_h[r][32 + kl];
        }
        __syncthreads();   // reads done before next iteration overwrites

        // c = x_proj fragment (f16 -> f32), contiguous b128 loads.
        v8f c[NTILE];
        const _Float16* xp = xp_base + (size_t)s * TILE_HALVES + lane * 8;
#pragma unroll
        for (int n = 0; n < NTILE; ++n) {
            const v8h xv = *(const v8h*)(xp + n * 256);
#pragma unroll
            for (int j = 0; j < 8; ++j) c[n][j] = (float)xv[j];
        }

        // c += h @ W_hh^T  (8 x v_wmma_f32_16x16x32_f16)
#pragma unroll
        for (int n = 0; n < NTILE; ++n) {
            c[n] = wmma_f16(a0, bw[n][0], c[n]);
            c[n] = wmma_f16(a1, bw[n][1], c[n]);
        }

        // h = tanh(c)  (v_tanh_f32 when available)
#pragma unroll
        for (int n = 0; n < NTILE; ++n)
#pragma unroll
            for (int j = 0; j < 8; ++j) h[n][j] = fast_tanh(c[n][j]);
    }

    // Final FC + sigmoid: lay h out row-major in LDS, lanes 0..15 reduce.
#pragma unroll
    for (int n = 0; n < NTILE; ++n) {
        const int col = n * 16 + r;
#pragma unroll
        for (int j = 0; j < 8; ++j)
            lds_hf[j + hi * 8][col] = h[n][j];
    }
    __syncthreads();

    if (lane < 16) {
        float acc = fc_b[0];
#pragma unroll
        for (int k = 0; k < H_DIM; ++k) acc += lds_hf[lane][k] * fc_w[k];
        out[b_tile * 16 + lane] = fast_sigmoid(acc);
    }
}

// ---------------------------------------------------------------------------
// Host launch.  Input order per setup_inputs():
//   0:x(int) 1:emb 2:W_ih 3:W_hh 4:b_ih 5:b_hh 6:fc_w 7:fc_b
// Workspace: 64 row-tiles * 512 steps * 1024 f16 = 64 MiB of d_ws.
// ---------------------------------------------------------------------------
extern "C" void kernel_launch(void* const* d_in, const int* in_sizes, int n_in,
                              void* d_out, int out_size, void* d_ws, size_t ws_size,
                              hipStream_t stream) {
    const int*   x    = (const int*)  d_in[0];
    const float* emb  = (const float*)d_in[1];
    const float* W_ih = (const float*)d_in[2];
    const float* W_hh = (const float*)d_in[3];
    const float* b_ih = (const float*)d_in[4];
    const float* b_hh = (const float*)d_in[5];
    const float* fc_w = (const float*)d_in[6];
    const float* fc_b = (const float*)d_in[7];
    float*       out  = (float*)d_out;
    _Float16*    xws  = (_Float16*)d_ws;

    (void)in_sizes; (void)n_in; (void)out_size; (void)ws_size;

    rnn_xproj_kernel<<<(B_SZ / 16) * S_LEN, 32, 0, stream>>>(x, emb, W_ih, b_ih, b_hh, xws);
    rnn_scan_kernel<<<B_SZ / 16, 32, 0, stream>>>(xws, W_hh, fc_w, fc_b, out);
}
